// NR_GraphAttention_46986942218773
// MI455X (gfx1250) — matrix operand order repaired and checked by reading, not compiled, not run on hardware
//
#include <hip/hip_runtime.h>
#include <hip/hip_bf16.h>

// Sizes fixed by the reference
#define N_NODES 50000
#define DDIM    256
#define ESZ     300000
#define KSEL_   150000
#define NPROXY_ 128
#define DC      768   // D*(DEPTH+1)

typedef __bf16 bf16t;
typedef __attribute__((ext_vector_type(16))) __bf16 v16bf;
typedef __attribute__((ext_vector_type(8)))  __bf16 v8bf;
typedef __attribute__((ext_vector_type(8)))  float  v8f;
typedef __attribute__((ext_vector_type(4))) unsigned int u32x4;
typedef __attribute__((ext_vector_type(8))) int i32x8;
typedef __attribute__((ext_vector_type(4))) int i32x4;

__device__ __forceinline__ v8f wmma_bf16(v16bf a, v16bf b, v8f c){
  return __builtin_amdgcn_wmma_f32_16x16x32_bf16(false, a, false, b, (short)0, c, false, false);
}
__device__ __forceinline__ v16bf cat8(v8bf lo, v8bf hi){
  return __builtin_shufflevector(lo, hi, 0,1,2,3,4,5,6,7,8,9,10,11,12,13,14,15);
}
__device__ __forceinline__ float wave_sum(float v){
  #pragma unroll
  for (int m = 16; m >= 1; m >>= 1) v += __shfl_xor(v, m, 32);
  return v;
}
__device__ __forceinline__ float wave_max(float v){
  #pragma unroll
  for (int m = 16; m >= 1; m >>= 1) v = fmaxf(v, __shfl_xor(v, m, 32));
  return v;
}
__device__ __forceinline__ int lbound(const int* __restrict__ a, int n, int key){
  int lo = 0, hi = n;
  while (lo < hi){ int mid = (lo + hi) >> 1; if (a[mid] < key) lo = mid + 1; else hi = mid; }
  return lo;
}

// ---- prep: bf16 transposes of B matrices (BT[n][k] layout for WMMA) ----
__global__ void k_prep(const float* __restrict__ w_keys, const float* __restrict__ gatek,
                       const float* __restrict__ proxy,
                       bf16t* __restrict__ wkT, bf16t* __restrict__ gateT, bf16t* __restrict__ proxyT){
  int i = blockIdx.x * 256 + threadIdx.x;
  if (i < 2 * 256 * 256){                      // w_keys[l][k][n] -> wkT[l][n][k]
    int l = i >> 16; int rem = i & 65535; int k = rem >> 8; int n = rem & 255;
    wkT[l * 65536 + n * 256 + k] = (bf16t)w_keys[i];
  }
  if (i < 768 * 768){                          // gate_kernel[k][n] -> gateT[n][k]
    int k = i / 768, n = i % 768;
    gateT[n * 768 + k] = (bf16t)gatek[i];
  }
  if (i < 128 * 768){                          // proxy[r][n] -> proxyT[n][r]
    int r = i / 768, n = i % 768;
    proxyT[n * 128 + r] = (bf16t)proxy[i];
  }
}

// ---- l2-normalized proxy rows (B for the attention GEMM: BT == proxyn) ----
__global__ void __launch_bounds__(32) k_proxy_norm(const float* __restrict__ proxy, bf16t* __restrict__ proxyn){
  int r = blockIdx.x, lane = threadIdx.x;
  const float* row = proxy + (size_t)r * DC;
  float v[24], ss = 0.f;
  #pragma unroll
  for (int j = 0; j < 24; j++){ v[j] = row[j * 32 + lane]; ss += v[j] * v[j]; }
  ss = wave_sum(ss);
  float inv = rsqrtf(fmaxf(ss, 1e-12f));
  #pragma unroll
  for (int j = 0; j < 24; j++) proxyn[(size_t)r * DC + j * 32 + lane] = (bf16t)(v[j] * inv);
}

// ---- node init: tanh(p), feats0 = features*tanh_p into out[:,0:256], copy p tail ----
__global__ void k_node_init(const float* __restrict__ features, const float* __restrict__ p,
                            float* __restrict__ tanh_p, float* __restrict__ out){
  int i = blockIdx.x * 256 + threadIdx.x;
  if (i >= N_NODES * DDIM) return;
  int n = i >> 8, d = i & 255;
  float t = tanhf(p[n]);
  out[(size_t)n * DC + d] = features[i] * t;
  if (d == 0){ tanh_p[n] = t; out[(size_t)N_NODES * DC + n] = p[n]; }
}

// ---- per layer: rels = l2n(rel_vals * rel_emb[rel_ids])  (bf16, wave per edge) ----
__global__ void k_rels_build(const float* __restrict__ rel_emb, const float* __restrict__ rel_vals,
                             const int* __restrict__ rel_ids, bf16t* __restrict__ rels){
  int w = (blockIdx.x * blockDim.x + threadIdx.x) >> 5;
  int lane = threadIdx.x & 31;
  if (w >= ESZ) return;
  int rid = rel_ids[w];
  float rv = rel_vals[w];
  v8f x = *(const v8f*)(rel_emb + (size_t)rid * DDIM + lane * 8);
  float ss = 0.f;
  #pragma unroll
  for (int j = 0; j < 8; j++){ x[j] *= rv; ss += x[j] * x[j]; }
  ss = wave_sum(ss);
  float inv = rsqrtf(fmaxf(ss, 1e-12f));
  v8bf o;
  #pragma unroll
  for (int j = 0; j < 8; j++) o[j] = (bf16t)(x[j] * inv);
  *(v8bf*)(rels + (size_t)w * DDIM + lane * 8) = o;
}

// ---- W_orth GEMM: src = rels[new_idx] @ W, wave computes 16x64 tile, K=256 ----
__global__ void __launch_bounds__(32) k_worth(const bf16t* __restrict__ rels, const bf16t* __restrict__ wkT_l,
                                              const int* __restrict__ new_idx, bf16t* __restrict__ srcbuf){
  int it = blockIdx.x, nc = blockIdx.y;
  int lane = threadIdx.x, m = lane & 15, h = lane >> 4;
  const bf16t* arow = rels + (size_t)new_idx[it * 16 + m] * DDIM;
  const bf16t* b0 = wkT_l + (size_t)(nc * 64 +  0 + m) * DDIM;
  const bf16t* b1 = wkT_l + (size_t)(nc * 64 + 16 + m) * DDIM;
  const bf16t* b2 = wkT_l + (size_t)(nc * 64 + 32 + m) * DDIM;
  const bf16t* b3 = wkT_l + (size_t)(nc * 64 + 48 + m) * DDIM;
  v8f c0 = {}, c1 = {}, c2 = {}, c3 = {};
  #pragma unroll
  for (int k0 = 0; k0 < 256; k0 += 32){
    __builtin_prefetch(arow + k0 + 64, 0, 0);
    v16bf a = cat8(*(const v8bf*)(arow + k0 + h * 16), *(const v8bf*)(arow + k0 + h * 16 + 8));
    c0 = wmma_bf16(a, *(const v16bf*)(b0 + k0 + h * 16), c0);
    c1 = wmma_bf16(a, *(const v16bf*)(b1 + k0 + h * 16), c1);
    c2 = wmma_bf16(a, *(const v16bf*)(b2 + k0 + h * 16), c2);
    c3 = wmma_bf16(a, *(const v16bf*)(b3 + k0 + h * 16), c3);
  }
  #pragma unroll
  for (int r = 0; r < 8; r++){
    size_t base = (size_t)(it * 16 + r + 8 * h) * DDIM + nc * 64 + m;
    srcbuf[base +  0] = (bf16t)c0[r];
    srcbuf[base + 16] = (bf16t)c1[r];
    srcbuf[base + 32] = (bf16t)c2[r];
    srcbuf[base + 48] = (bf16t)c3[r];
  }
}

// ---- scatter src rows back into rels (separate pass: reads use pre-update values) ----
__global__ void k_scatter(const bf16t* __restrict__ srcbuf, const int* __restrict__ new_idx,
                          bf16t* __restrict__ rels){
  int w = (blockIdx.x * blockDim.x + threadIdx.x) >> 5;
  int lane = threadIdx.x & 31;
  if (w >= KSEL_) return;
  int row = new_idx[w];
  *(v8bf*)(rels + (size_t)row * DDIM + lane * 8) =
      *(const v8bf*)(srcbuf + (size_t)w * DDIM + lane * 8);
}

// ---- edge compute: scale by tanh_p[col], Householder reflect, attention logit ----
__global__ void k_edge(const bf16t* __restrict__ rels, const float* __restrict__ out,
                       const float* __restrict__ tanh_p, const float* __restrict__ attn_l,
                       const int* __restrict__ adj_col, bf16t* __restrict__ neighs,
                       float* __restrict__ att, int lcol){
  int e = (blockIdx.x * blockDim.x + threadIdx.x) >> 5;
  int lane = threadIdx.x & 31;
  if (e >= ESZ) return;
  int col = adj_col[e];
  float tp = tanh_p[col];
  v8bf rb = *(const v8bf*)(rels + (size_t)e * DDIM + lane * 8);
  v8f  nv = *(const v8f*)(out + (size_t)col * DC + lcol + lane * 8);
  const float* ak = attn_l + lane * 8;
  float r[8], dot = 0.f, al = 0.f;
  #pragma unroll
  for (int j = 0; j < 8; j++){
    r[j] = (float)rb[j] * tp;
    dot += nv[j] * r[j];
    al  += r[j] * ak[j];
  }
  dot = wave_sum(dot);
  al  = wave_sum(al);
  v8bf o;
  #pragma unroll
  for (int j = 0; j < 8; j++) o[j] = (bf16t)(nv[j] - 2.f * dot * r[j]);
  *(v8bf*)(neighs + (size_t)e * DDIM + lane * 8) = o;
  if (lane == 0) att[e] = al;
}

// ---- per-row segment softmax + weighted aggregate (wave per row, adj_row sorted) ----
__global__ void k_rowagg(const bf16t* __restrict__ neighs, const float* __restrict__ att,
                         const int* __restrict__ adj_row, const float* __restrict__ tanh_p,
                         float* __restrict__ out, int lcol_out){
  int n = (blockIdx.x * blockDim.x + threadIdx.x) >> 5;
  int lane = threadIdx.x & 31;
  if (n >= N_NODES) return;
  int lo = lbound(adj_row, ESZ, n);
  int hi = lbound(adj_row, ESZ, n + 1);
  float* dst = out + (size_t)n * DC + lcol_out + lane * 8;
  if (lo >= hi){ v8f z = {}; *(v8f*)dst = z; return; }
  float mx = -3.4e38f;
  for (int e = lo + lane; e < hi; e += 32) mx = fmaxf(mx, att[e]);
  mx = wave_max(mx);
  float acc[8] = {0,0,0,0,0,0,0,0};
  float s = 0.f;
  for (int e = lo; e < hi; ++e){
    float wgt = __expf(att[e] - mx);
    s += wgt;
    v8bf nb = *(const v8bf*)(neighs + (size_t)e * DDIM + lane * 8);
    #pragma unroll
    for (int j = 0; j < 8; j++) acc[j] += wgt * (float)nb[j];
  }
  float scale = tanh_p[n] / s;
  v8f o;
  #pragma unroll
  for (int j = 0; j < 8; j++) o[j] = acc[j] * scale;
  *(v8f*)dst = o;
}

// ---- l2n(out) -> bf16 (A matrix of proxy-attention GEMM) ----
__global__ void k_outnorm(const float* __restrict__ out, bf16t* __restrict__ outn){
  int n = (blockIdx.x * blockDim.x + threadIdx.x) >> 5;
  int lane = threadIdx.x & 31;
  if (n >= N_NODES) return;
  const float* row = out + (size_t)n * DC;
  float v[24], ss = 0.f;
  #pragma unroll
  for (int j = 0; j < 24; j++){ v[j] = row[j * 32 + lane]; ss += v[j] * v[j]; }
  ss = wave_sum(ss);
  float inv = rsqrtf(fmaxf(ss, 1e-12f));
  #pragma unroll
  for (int j = 0; j < 24; j++) outn[(size_t)n * DC + j * 32 + lane] = (bf16t)(v[j] * inv);
}

// ---- fused: proxy_att GEMM (16x128) -> softmax (half-wave) -> pf = out - att@proxy ----
__global__ void __launch_bounds__(32) k_proxy_fused(const bf16t* __restrict__ outn,
    const bf16t* __restrict__ proxyn, const bf16t* __restrict__ proxyT,
    const float* __restrict__ out, float* __restrict__ pf){
  __shared__ float lsA[16 * 128];
  int it = blockIdx.x;
  int lane = threadIdx.x, m = lane & 15, h = lane >> 4;
  const bf16t* arow = outn + (size_t)(it * 16 + m) * DC;
  v8f acc[8] = {};
  for (int k0 = 0; k0 < DC; k0 += 32){
    v16bf a = cat8(*(const v8bf*)(arow + k0 + h * 16), *(const v8bf*)(arow + k0 + h * 16 + 8));
    #pragma unroll
    for (int t = 0; t < 8; t++){
      v16bf b = *(const v16bf*)(proxyn + (size_t)(t * 16 + m) * DC + k0 + h * 16);
      acc[t] = wmma_bf16(a, b, acc[t]);
    }
  }
  // softmax over 128 cols; row r+8h owned by lane-half h; reduce across 16 lanes of half
  #pragma unroll
  for (int r = 0; r < 8; r++){
    float mx = -3.4e38f;
    #pragma unroll
    for (int t = 0; t < 8; t++) mx = fmaxf(mx, acc[t][r]);
    #pragma unroll
    for (int d = 8; d >= 1; d >>= 1) mx = fmaxf(mx, __shfl_xor(mx, d, 32));
    float s = 0.f;
    #pragma unroll
    for (int t = 0; t < 8; t++){ float e = __expf(acc[t][r] - mx); acc[t][r] = e; s += e; }
    #pragma unroll
    for (int d = 8; d >= 1; d >>= 1) s += __shfl_xor(s, d, 32);
    float inv = 1.f / s;
    #pragma unroll
    for (int t = 0; t < 8; t++) lsA[(r + 8 * h) * 128 + t * 16 + m] = acc[t][r] * inv;
  }
  __syncthreads();
  // pf = out - proxy_att @ proxy   (A restaged from LDS, K=128, N=768 in 64-col chunks)
  for (int ncb = 0; ncb < 12; ++ncb){
    v8f c[4] = {};
    #pragma unroll
    for (int k0 = 0; k0 < 128; k0 += 32){
      v16bf a;
      #pragma unroll
      for (int j = 0; j < 8; j++){
        a[j]     = (bf16t)lsA[m * 128 + k0 + h * 16 + j];
        a[8 + j] = (bf16t)lsA[m * 128 + k0 + h * 16 + 8 + j];
      }
      #pragma unroll
      for (int t = 0; t < 4; t++){
        v16bf b = *(const v16bf*)(proxyT + (size_t)(ncb * 64 + t * 16 + m) * NPROXY_ + k0 + h * 16);
        c[t] = wmma_bf16(a, b, c[t]);
      }
    }
    #pragma unroll
    for (int t = 0; t < 4; t++)
      #pragma unroll
      for (int r = 0; r < 8; r++){
        size_t idx = (size_t)(it * 16 + r + 8 * h) * DC + ncb * 64 + t * 16 + m;
        pf[idx] = out[idx] - c[t][r];
      }
  }
}

// ---- gate GEMM (K=768) + final blend, cooperative 8-wave workgroup ----
// Workgroup computes a 128-row x 64-col tile. The 64x128 bf16 B-chunk of gateT
// is DMA'd into LDS once per workgroup by wave 0 via the Tensor Data Mover
// (tensor_load_to_lds + s_wait_tensorcnt), then shared by all 8 waves.
__global__ void __launch_bounds__(256) k_gate(const float* __restrict__ pf,
    const bf16t* __restrict__ gateT, float* __restrict__ out){
  __shared__ bf16t lsB[64 * 128];   // only LDS allocation -> descriptor lds_addr = 0
  int wid  = threadIdx.x >> 5;
  int lane = threadIdx.x & 31, m = lane & 15, h = lane >> 4;
  int it = blockIdx.x * 8 + wid;            // 16-row tile index (0..3127, N has 3125)
  int nc = blockIdx.y;                      // 64-col chunk
  int ar = it * 16 + m;
  if (ar >= N_NODES) ar = N_NODES - 1;      // clamp: keep EXEC all-ones for WMMA
  const float* arow = pf + (size_t)ar * DC;
  v8f c[4] = {};
  for (int kc = 0; kc < 6; ++kc){           // 6 K-chunks of 128
    if (wid == 0){
      unsigned long long ga =
          (unsigned long long)(uintptr_t)(gateT + (size_t)nc * 64 * DC + kc * 128);
      // D# group0: count=1 | lds_addr=0 | global_addr | type=2 (ISA 8.3)
      u32x4 g0 = { 1u, 0u,
                   (unsigned)(ga & 0xFFFFFFFFull),
                   (unsigned)((ga >> 32) & 0x01FFFFFFull) | (2u << 30) };
      // D# group1: data_size=2B; tensor_dim0=768; tensor_dim1=768;
      //            tile_dim0=128; tile_dim1=64; tensor_dim0_stride=768 (ISA 8.4)
      i32x8 g1 = { 0x00010000,
                   (int)((DC & 0xFFFF) << 16),
                   (int)((DC & 0xFFFF) << 16),
                   (int)(128 << 16),
                   64,
                   DC,
                   0, 0 };
      i32x4 gz4 = { 0, 0, 0, 0 };
      i32x8 gz8 = { 0, 0, 0, 0, 0, 0, 0, 0 };
      __builtin_amdgcn_tensor_load_to_lds(g0, g1, gz4, gz4, gz8, 0);
      __builtin_amdgcn_s_wait_tensorcnt(0);
    }
    __syncthreads();
    #pragma unroll
    for (int k0 = 0; k0 < 128; k0 += 32){
      int kg = kc * 128 + k0;
      v8f flo = *(const v8f*)(arow + kg + h * 16);
      v8f fhi = *(const v8f*)(arow + kg + h * 16 + 8);
      v16bf a;
      #pragma unroll
      for (int j = 0; j < 8; j++){ a[j] = (bf16t)flo[j]; a[8 + j] = (bf16t)fhi[j]; }
      #pragma unroll
      for (int t = 0; t < 4; t++){
        v16bf b = *(const v16bf*)(lsB + (t * 16 + m) * 128 + k0 + h * 16);
        c[t] = wmma_bf16(a, b, c[t]);
      }
    }
    __syncthreads();                         // tile consumed before next DMA
  }
  #pragma unroll
  for (int t = 0; t < 4; t++)
    #pragma unroll
    for (int r = 0; r < 8; r++){
      int row = it * 16 + r + 8 * h;
      if (row < N_NODES){
        size_t idx = (size_t)row * DC + nc * 64 + t * 16 + m;
        float g = 1.f / (1.f + __expf(-c[t][r]));
        out[idx] = g * out[idx] + (1.f - g) * pf[idx];
      }
    }
}

extern "C" void kernel_launch(void* const* d_in, const int* in_sizes, int n_in,
                              void* d_out, int out_size, void* d_ws, size_t ws_size,
                              hipStream_t stream) {
  (void)in_sizes; (void)n_in; (void)out_size; (void)ws_size;
  const float* features = (const float*)d_in[0];
  const float* rel_emb  = (const float*)d_in[1];
  const float* p        = (const float*)d_in[2];
  const float* w_keys   = (const float*)d_in[3];
  const float* attn     = (const float*)d_in[4];
  const float* proxy    = (const float*)d_in[5];
  const float* gatek    = (const float*)d_in[6];
  const float* rel_vals = (const float*)d_in[7];
  const int*   adj_row  = (const int*)d_in[8];
  const int*   adj_col  = (const int*)d_in[9];
  const int*   rel_ids  = (const int*)d_in[10];
  const int*   new_idx  = (const int*)d_in[11];
  float* out = (float*)d_out;

  // workspace layout (all 256B aligned); ~541 MB total
  char* wsp = (char*)d_ws;
  auto take = [&](size_t bytes) -> char* {
    char* r = wsp; wsp += (bytes + 255) & ~(size_t)255; return r;
  };
  float* tanh_p = (float*)take((size_t)N_NODES * 4);
  bf16t* rels   = (bf16t*)take((size_t)ESZ * DDIM * 2);
  bf16t* neighs = (bf16t*)take((size_t)ESZ * DDIM * 2);
  bf16t* srcbuf = neighs;  // alias: srcbuf dead before neighs is written each layer
  float* att    = (float*)take((size_t)ESZ * 4);
  bf16t* wkT    = (bf16t*)take((size_t)2 * 256 * 256 * 2);
  bf16t* gateT  = (bf16t*)take((size_t)768 * 768 * 2);
  bf16t* proxyn = (bf16t*)take((size_t)128 * 768 * 2);
  bf16t* proxyT = (bf16t*)take((size_t)768 * 128 * 2);
  bf16t* outn   = (bf16t*)take((size_t)N_NODES * DC * 2);
  float* pf     = (float*)take((size_t)N_NODES * DC * 4);

  k_prep<<<2304, 256, 0, stream>>>(w_keys, gatek, proxy, wkT, gateT, proxyT);
  k_proxy_norm<<<128, 32, 0, stream>>>(proxy, proxyn);
  k_node_init<<<(N_NODES * DDIM + 255) / 256, 256, 0, stream>>>(features, p, tanh_p, out);

  for (int l = 0; l < 2; ++l){
    k_rels_build<<<ESZ / 8, 256, 0, stream>>>(rel_emb, rel_vals, rel_ids, rels);
    k_worth<<<dim3(KSEL_ / 16, 4), 32, 0, stream>>>(rels, wkT + l * 65536, new_idx, srcbuf);
    k_scatter<<<(KSEL_ + 7) / 8, 256, 0, stream>>>(srcbuf, new_idx, rels);
    k_edge<<<ESZ / 8, 256, 0, stream>>>(rels, out, tanh_p, attn + l * 256, adj_col, neighs, att, 256 * l);
    k_rowagg<<<(N_NODES + 7) / 8, 256, 0, stream>>>(neighs, att, adj_row, tanh_p, out, 256 * (l + 1));
  }

  k_outnorm<<<(N_NODES + 7) / 8, 256, 0, stream>>>(out, outn);
  k_proxy_fused<<<N_NODES / 16, 32, 0, stream>>>(outn, proxyn, proxyT, out, pf);
  // 3125 row tiles of 16 -> 391 workgroups of 8 waves; 12 column chunks of 64
  k_gate<<<dim3((N_NODES / 16 + 7) / 8, 12), 256, 0, stream>>>(pf, gateT, out);
}
